// SE3Transformer_73409581023953
// MI455X (gfx1250) — matrix operand order repaired and checked by reading, not compiled, not run on hardware
//
#include <hip/hip_runtime.h>
#include <math.h>

// ---------------- problem constants (from reference) ----------------
#define N_NODES 20000
#define K_EDGES 8
#define E_EDGES (N_NODES * K_EDGES)   // 160000
#define NBASIS  16
#define HDIM    64
#define A0c 16
#define A1c 8
#define Q0c 8
#define Q1c 4
#define B0c 16
#define B1c 8
#define RMAXc 3.5f
#define F_COLS  40                    // A0 + 3*A1
#define WK_COLS 288                   // A0*Q0 + A1*Q0 + A0*Q1 + A1*Q1
#define WV_COLS 576                   // A0*B0 + A1*B0 + A0*B1 + A1*B1
#define OUT_COLS 40                   // B0 + 3*B1

#define BLK_EDGES 64                  // edges per block (= 4 WMMA M-tiles, 8 nodes)
#define BLK_THREADS 128               // 4 waves

// WMMA vector types (gfx1250, wave32)
typedef __attribute__((ext_vector_type(16))) __bf16 v16bf;
typedef __attribute__((ext_vector_type(8)))  float  v8f;

// ---------------- helpers ----------------
__device__ __forceinline__ unsigned short f2bf_bits(float x) {
  union { float f; unsigned u; } c; c.f = x;
  unsigned r = c.u + 0x7FFFu + ((c.u >> 16) & 1u);   // round-to-nearest-even
  return (unsigned short)(r >> 16);
}
__device__ __forceinline__ __bf16 f2bf(float x) {
  union { unsigned short s; __bf16 h; } o; o.s = f2bf_bits(x); return o.h;
}

__device__ __forceinline__ v8f wmma_bf16(v16bf a, v16bf b, v8f c) {
  // D = A(16x32) * B(32x16) + C ; f32 accumulate
  return __builtin_amdgcn_wmma_f32_16x16x32_bf16(false, a, false, b, (short)0, c,
                                                 false, false);
}

// A-operand fragment from an LDS row (documented 16-bit A 16x32 layout):
// lane L: M = L%16, hi = L/16 ; halves j<8 -> K = kbase + hi*8 + j,
//                               halves j>=8 -> K = kbase + 16 + hi*8 + (j-8)
__device__ __forceinline__ v16bf load_a_frag(const __bf16* row, int hi, int kbase) {
  v16bf a;
  const __bf16* p0 = row + kbase + hi * 8;
  const __bf16* p1 = p0 + 16;
#pragma unroll
  for (int j = 0; j < 8; ++j) { a[j] = p0[j]; a[8 + j] = p1[j]; }
  return a;
}

// ---------------- weight packing into B-fragment order ----------------
// B fragment (32x16 K x N): lane L holds column n = L%16; half j holds
// K = ks*32 + (j&7) + 8*(L/16) + 16*(j>>3).  Fragment index = nt*kSteps + ks, 512 halves each.
__global__ __launch_bounds__(256) void se3_pack_bfrag(
    const float* __restrict__ src, unsigned short* __restrict__ dst,
    int Ksrc, int Ncols, int nTiles, int kSteps, float scale) {
  int idx = blockIdx.x * blockDim.x + threadIdx.x;
  int total = nTiles * kSteps * 512;
  if (idx >= total) return;
  int frag   = idx >> 9;
  int within = idx & 511;
  int lane = within >> 4;
  int j    = within & 15;
  int nt = frag / kSteps;
  int ks = frag - nt * kSteps;
  int n  = lane & 15;
  int hi = lane >> 4;
  int k  = ks * 32 + (j & 7) + hi * 8 + (j >> 3) * 16;
  float v = (k < Ksrc) ? src[k * Ncols + nt * 16 + n] * scale : 0.0f;
  dst[idx] = f2bf_bits(v);
}

// ---------------- per-node query projections ----------------
__global__ __launch_bounds__(256) void se3_qproj(
    const float* __restrict__ f, const float* __restrict__ Wq0,
    const float* __restrict__ Wq1, float* __restrict__ q0,
    float* __restrict__ q1) {
  int n = blockIdx.x * blockDim.x + threadIdx.x;
  if (n >= N_NODES) return;
  const float* fr = f + (size_t)n * F_COLS;
  float f0[A0c], f1[A1c][3];
#pragma unroll
  for (int a = 0; a < A0c; ++a) f0[a] = fr[a];
#pragma unroll
  for (int a = 0; a < A1c; ++a)
#pragma unroll
    for (int i = 0; i < 3; ++i) f1[a][i] = fr[A0c + a * 3 + i];

  const float s0 = 0.25f;                  // 1/sqrt(A0)
  const float s1 = 0.35355339059327373f;   // 1/sqrt(A1)
#pragma unroll
  for (int b = 0; b < Q0c; ++b) {
    float acc = 0.f;
#pragma unroll
    for (int a = 0; a < A0c; ++a) acc += f0[a] * Wq0[a * Q0c + b];
    q0[n * Q0c + b] = acc * s0;
  }
#pragma unroll
  for (int q = 0; q < Q1c; ++q)
#pragma unroll
    for (int i = 0; i < 3; ++i) {
      float acc = 0.f;
#pragma unroll
      for (int a = 0; a < A1c; ++a) acc += f1[a][i] * Wq1[a * Q1c + q];
      q1[n * (Q1c * 3) + q * 3 + i] = acc * s1;
    }
}

// ---------------- main fused edge kernel ----------------
// One block = 128 threads (4 waves) = 64 edges (4 WMMA M-tiles) = 8 dst nodes.
// Each weight B-fragment is loaded from L2 exactly once per block and applied
// to all 4 M-tiles (A fragments held in registers); the nt (column-tile) range
// is strided across the 4 waves.
__global__ __launch_bounds__(BLK_THREADS) void se3_edge_main(
    const float* __restrict__ f, const float* __restrict__ pos,
    const int* __restrict__ edge_src,
    const float* __restrict__ Wd0, const float* __restrict__ Wd1,
    const float* __restrict__ q0buf, const float* __restrict__ q1buf,
    const unsigned short* __restrict__ pW1k, const unsigned short* __restrict__ pW2k,
    const unsigned short* __restrict__ pW1v, const unsigned short* __restrict__ pW2v,
    float* __restrict__ out) {
  __shared__ __bf16 sBas[BLK_EDGES][32];        // basis, K padded to 32   (4 KB)
  __shared__ __bf16 sH[BLK_EDGES][HDIM];        // hidden, k then v pass   (8 KB)
  __shared__ float  sWK[BLK_EDGES][WK_COLS];    // 72 KB
  __shared__ float  sWV[BLK_EDGES][WV_COLS];    // 144 KB
  __shared__ float  sY1[BLK_EDGES][3];
  __shared__ float  sCut[BLK_EDGES];
  __shared__ float  sLog[BLK_EDGES];
  __shared__ float  sWe[BLK_EDGES];
  __shared__ float  sV[BLK_EDGES][OUT_COLS];    // 10 KB

  const int tid   = threadIdx.x;
  const int lane  = tid & 31;
  const int wave  = tid >> 5;                   // 0..3
  const int ebase = blockIdx.x * BLK_EDGES;

  // ---- phase A: per-edge geometry + radial basis (threads 0..63) ----
  if (tid < BLK_EDGES) {
    const int e   = ebase + tid;
    const int src = edge_src[e];
    const int dst = e >> 3;              // edge_dst = repeat(arange(N), 8)
    float ev[3];
#pragma unroll
    for (int i = 0; i < 3; ++i) ev[i] = pos[src * 3 + i] - pos[dst * 3 + i];
    float r = sqrtf(ev[0]*ev[0] + ev[1]*ev[1] + ev[2]*ev[2] + 1e-12f);
    const float step   = RMAXc / 17.0f;          // RMAX/(NB+1)
    const float istep  = 1.0f / step;
    const float bscale = 4.0f / 1.12f;           // sqrt(NB)/1.12
#pragma unroll
    for (int c = 0; c < NBASIS; ++c) {
      float d = (r - step * (float)(c + 1)) * istep;
      sBas[tid][c] = f2bf(__expf(-d * d) * bscale);
    }
#pragma unroll
    for (int c = NBASIS; c < 32; ++c) sBas[tid][c] = f2bf(0.0f);
    float xcut = 10.0f * (1.0f - r / RMAXc);
    sCut[tid] = (xcut > 0.0f) ? __expf(-1.0f / xcut) : 0.0f;
    float inv_r = 1.0f / r;
    const float s3 = 1.7320508075688772f;
#pragma unroll
    for (int i = 0; i < 3; ++i) sY1[tid][i] = s3 * ev[i] * inv_r;
  }
  __syncthreads();

  // ---- phase B: WMMA MLPs (all waves, EXEC all-ones) ----
  const int m  = lane & 15;
  const int hi = lane >> 4;
  const int rowC = (hi << 3);                   // C/D layout: row = v + 8*hi
  const v16bf* BW1k = reinterpret_cast<const v16bf*>(pW1k);
  const v16bf* BW2k = reinterpret_cast<const v16bf*>(pW2k);
  const v16bf* BW1v = reinterpret_cast<const v16bf*>(pW1v);
  const v16bf* BW2v = reinterpret_cast<const v16bf*>(pW2v);

  // A fragments (basis) for all 4 M-tiles, kept in registers
  v16bf aBas[4];
#pragma unroll
  for (int mt = 0; mt < 4; ++mt)
    aBas[mt] = load_a_frag(&sBas[mt * 16 + m][0], hi, 0);

  // GEMM1-k: hidden = silu(basis @ W1k); wave w owns column tile nt = w
  {
    const int nt = wave;                        // 4 tiles, one per wave
    v16bf b = BW1k[nt * 32 + lane];
#pragma unroll
    for (int mt = 0; mt < 4; ++mt) {
      v8f acc = {};
      acc = wmma_bf16(aBas[mt], b, acc);
#pragma unroll
      for (int v = 0; v < 8; ++v) {
        float x = acc[v];
        sH[mt * 16 + rowC + v][nt * 16 + m] = f2bf(x / (1.0f + __expf(-x)));
      }
    }
  }
  __syncthreads();

  // GEMM2-k: wk = h @ W2k ; 18 column tiles strided over the 4 waves
  {
    v16bf aH[4][2];
#pragma unroll
    for (int mt = 0; mt < 4; ++mt) {
      aH[mt][0] = load_a_frag(&sH[mt * 16 + m][0], hi, 0);
      aH[mt][1] = load_a_frag(&sH[mt * 16 + m][0], hi, 32);
    }
    for (int nt = wave; nt < 18; nt += 4) {
      v16bf b0 = BW2k[(nt * 2 + 0) * 32 + lane];
      v16bf b1 = BW2k[(nt * 2 + 1) * 32 + lane];
#pragma unroll
      for (int mt = 0; mt < 4; ++mt) {
        v8f acc = {};
        acc = wmma_bf16(aH[mt][0], b0, acc);
        acc = wmma_bf16(aH[mt][1], b1, acc);
#pragma unroll
        for (int v = 0; v < 8; ++v)
          sWK[mt * 16 + rowC + v][nt * 16 + m] = acc[v];
      }
    }
  }
  __syncthreads();   // all waves done reading sH before the v-pass overwrites it

  // GEMM1-v
  {
    const int nt = wave;
    v16bf b = BW1v[nt * 32 + lane];
#pragma unroll
    for (int mt = 0; mt < 4; ++mt) {
      v8f acc = {};
      acc = wmma_bf16(aBas[mt], b, acc);
#pragma unroll
      for (int v = 0; v < 8; ++v) {
        float x = acc[v];
        sH[mt * 16 + rowC + v][nt * 16 + m] = f2bf(x / (1.0f + __expf(-x)));
      }
    }
  }
  __syncthreads();

  // GEMM2-v: wv = h @ W2v ; 36 column tiles strided over the 4 waves
  {
    v16bf aH[4][2];
#pragma unroll
    for (int mt = 0; mt < 4; ++mt) {
      aH[mt][0] = load_a_frag(&sH[mt * 16 + m][0], hi, 0);
      aH[mt][1] = load_a_frag(&sH[mt * 16 + m][0], hi, 32);
    }
    for (int nt = wave; nt < 36; nt += 4) {
      v16bf b0 = BW2v[(nt * 2 + 0) * 32 + lane];
      v16bf b1 = BW2v[(nt * 2 + 1) * 32 + lane];
#pragma unroll
      for (int mt = 0; mt < 4; ++mt) {
        v8f acc = {};
        acc = wmma_bf16(aH[mt][0], b0, acc);
        acc = wmma_bf16(aH[mt][1], b1, acc);
#pragma unroll
        for (int v = 0; v < 8; ++v)
          sWV[mt * 16 + rowC + v][nt * 16 + m] = acc[v];
      }
    }
  }
  __syncthreads();

  // ---- phase C: per-edge tensor products + logits, split over all threads:
  //      threads 0..63  -> k-path + logits for edge t
  //      threads 64..127-> v-path for edge t-64
  {
    const int t   = tid & (BLK_EDGES - 1);
    const int e   = ebase + t;
    const int src = edge_src[e];
    const float* fr = f + (size_t)src * F_COLS;
    float f0s[A0c], f1s[A1c][3];
#pragma unroll
    for (int a = 0; a < A0c; ++a) f0s[a] = fr[a];
#pragma unroll
    for (int a = 0; a < A1c; ++a)
#pragma unroll
      for (int i = 0; i < 3; ++i) f1s[a][i] = fr[A0c + a * 3 + i];
    float y1v[3] = { sY1[t][0], sY1[t][1], sY1[t][2] };
    const float inv_s3 = 0.5773502691896258f;
    float dot11[A1c];
#pragma unroll
    for (int a = 0; a < A1c; ++a)
      dot11[a] = (f1s[a][0]*y1v[0] + f1s[a][1]*y1v[1] + f1s[a][2]*y1v[2]) * inv_s3;
    const float tpn = 0.20412414523193154f;      // 1/sqrt(A0+A1)

    if (tid < BLK_EDGES) {
      // ---- k path: w00[0,128) w110[128,192) w01[192,256) w10[256,288) ----
      const float* wkr = sWK[t];
      float k0[Q0c];
#pragma unroll
      for (int b = 0; b < Q0c; ++b) {
        float acc = 0.f;
#pragma unroll
        for (int a = 0; a < A0c; ++a) acc += f0s[a] * wkr[a * Q0c + b];
#pragma unroll
        for (int a = 0; a < A1c; ++a) acc += dot11[a] * wkr[128 + a * Q0c + b];
        k0[b] = acc * tpn;
      }
      float k1[Q1c][3];
#pragma unroll
      for (int b = 0; b < Q1c; ++b) {
        float c01 = 0.f;
#pragma unroll
        for (int a = 0; a < A0c; ++a) c01 += f0s[a] * wkr[192 + a * Q1c + b];
#pragma unroll
        for (int i = 0; i < 3; ++i) {
          float acc = c01 * y1v[i];
#pragma unroll
          for (int a = 0; a < A1c; ++a) acc += f1s[a][i] * wkr[256 + a * Q1c + b];
          k1[b][i] = acc * tpn;
        }
      }
      // ---- logits ----
      const int dst = e >> 3;
      const float* q0d = q0buf + (size_t)dst * Q0c;
      const float* q1d = q1buf + (size_t)dst * (Q1c * 3);
      float l0 = 0.f;
#pragma unroll
      for (int a = 0; a < Q0c; ++a)
#pragma unroll
        for (int b = 0; b < Q0c; ++b) l0 += q0d[a] * Wd0[a * Q0c + b] * k0[b];
      float l1 = 0.f;
#pragma unroll
      for (int a = 0; a < Q1c; ++a)
#pragma unroll
        for (int b = 0; b < Q1c; ++b) {
          float dki = q1d[a*3]*k1[b][0] + q1d[a*3+1]*k1[b][1] + q1d[a*3+2]*k1[b][2];
          l1 += Wd1[a * Q1c + b] * dki;
        }
      const float dnorm = 0.11180339887498948f;  // 1/sqrt(Q0^2+Q1^2)
      sLog[t] = (l0 + l1 * inv_s3) * dnorm;
    } else {
      // ---- v path: w00[0,256) w110[256,384) w01[384,512) w10[512,576) ----
      const float* wvr = sWV[t];
#pragma unroll
      for (int b = 0; b < B0c; ++b) {
        float acc = 0.f;
#pragma unroll
        for (int a = 0; a < A0c; ++a) acc += f0s[a] * wvr[a * B0c + b];
#pragma unroll
        for (int a = 0; a < A1c; ++a) acc += dot11[a] * wvr[256 + a * B0c + b];
        sV[t][b] = acc * tpn;
      }
#pragma unroll
      for (int b = 0; b < B1c; ++b) {
        float c01 = 0.f;
#pragma unroll
        for (int a = 0; a < A0c; ++a) c01 += f0s[a] * wvr[384 + a * B1c + b];
#pragma unroll
        for (int i = 0; i < 3; ++i) {
          float acc = c01 * y1v[i];
#pragma unroll
          for (int a = 0; a < A1c; ++a) acc += f1s[a][i] * wvr[512 + a * B1c + b];
          sV[t][B0c + b * 3 + i] = acc * tpn;
        }
      }
    }
  }
  __syncthreads();

  // ---- phase D: per-node softmax * cutoff (8 nodes / block) ----
  if (tid < BLK_EDGES / 8) {
    int b8 = tid * 8;
    float mx = -3.4e38f;
#pragma unroll
    for (int j = 0; j < 8; ++j) mx = fmaxf(mx, sLog[b8 + j]);
    float z[8]; float s = 0.f;
#pragma unroll
    for (int j = 0; j < 8; ++j) { z[j] = __expf(sLog[b8 + j] - mx); s += z[j]; }
    float invs = 1.0f / s;
#pragma unroll
    for (int j = 0; j < 8; ++j) {
      float alpha = z[j] * invs * sCut[b8 + j];
      sWe[b8 + j] = sqrtf(fmaxf(alpha, 0.0f) + 1e-12f);
    }
  }
  __syncthreads();

  // ---- output reduction: 8 nodes x 40 channels ----
  for (int idx = tid; idx < (BLK_EDGES / 8) * OUT_COLS; idx += BLK_THREADS) {
    int nd = idx / OUT_COLS;
    int c  = idx - nd * OUT_COLS;
    float acc = 0.f;
#pragma unroll
    for (int j = 0; j < 8; ++j) acc += sWe[nd * 8 + j] * sV[nd * 8 + j][c];
    out[(size_t)(blockIdx.x * (BLK_EDGES / 8) + nd) * OUT_COLS + c] = acc;
  }
}

// ---------------- host launcher ----------------
extern "C" void kernel_launch(void* const* d_in, const int* in_sizes, int n_in,
                              void* d_out, int out_size, void* d_ws, size_t ws_size,
                              hipStream_t stream) {
  const float* f   = (const float*)d_in[0];
  const float* pos = (const float*)d_in[1];
  const float* W1k = (const float*)d_in[2];
  const float* W2k = (const float*)d_in[3];
  const float* W1v = (const float*)d_in[4];
  const float* W2v = (const float*)d_in[5];
  const float* Wq0 = (const float*)d_in[6];
  const float* Wq1 = (const float*)d_in[7];
  const float* Wd0 = (const float*)d_in[8];
  const float* Wd1 = (const float*)d_in[9];
  const int*  esrc = (const int*)d_in[11];
  float* out = (float*)d_out;

  char* ws = (char*)d_ws;
  size_t off = 0;
  auto take = [&](size_t bytes) {
    size_t cur = off;
    off += (bytes + 255) & ~(size_t)255;
    return cur;
  };
  float* q0            = (float*)(ws + take((size_t)N_NODES * Q0c * sizeof(float)));
  float* q1            = (float*)(ws + take((size_t)N_NODES * Q1c * 3 * sizeof(float)));
  unsigned short* pW1k = (unsigned short*)(ws + take((size_t)4  * 512 * 2));
  unsigned short* pW2k = (unsigned short*)(ws + take((size_t)36 * 512 * 2));
  unsigned short* pW1v = (unsigned short*)(ws + take((size_t)4  * 512 * 2));
  unsigned short* pW2v = (unsigned short*)(ws + take((size_t)72 * 512 * 2));

  // pack weights (scales 1/sqrt(NB), 1/sqrt(H) baked in)
  se3_pack_bfrag<<<(4 * 1 * 512 + 255) / 256, 256, 0, stream>>>(
      W1k, pW1k, NBASIS, HDIM, 4, 1, 0.25f);
  se3_pack_bfrag<<<(18 * 2 * 512 + 255) / 256, 256, 0, stream>>>(
      W2k, pW2k, HDIM, WK_COLS, 18, 2, 0.125f);
  se3_pack_bfrag<<<(4 * 1 * 512 + 255) / 256, 256, 0, stream>>>(
      W1v, pW1v, NBASIS, HDIM, 4, 1, 0.25f);
  se3_pack_bfrag<<<(36 * 2 * 512 + 255) / 256, 256, 0, stream>>>(
      W2v, pW2v, HDIM, WV_COLS, 36, 2, 0.125f);

  se3_qproj<<<(N_NODES + 255) / 256, 256, 0, stream>>>(f, Wq0, Wq1, q0, q1);

  se3_edge_main<<<E_EDGES / BLK_EDGES, BLK_THREADS, 0, stream>>>(
      f, pos, esrc, Wd0, Wd1, q0, q1, pW1k, pW2k, pW1v, pW2v, out);
}
